// WindowAttention_3788161155755
// MI455X (gfx1250) — matrix operand order repaired and checked by reading, not compiled, not run on hardware
//
#include <hip/hip_runtime.h>
#include <hip/hip_bf16.h>

typedef __attribute__((ext_vector_type(16))) _Float16 v16h;
typedef __attribute__((ext_vector_type(8)))  _Float16 v8h;
typedef __attribute__((ext_vector_type(4)))  _Float16 v4h;
typedef __attribute__((ext_vector_type(8)))  float    v8f;

#define DIMC   256
#define HEADS  8
#define HDIM   32
#define NTOK   64
#define NWIN   4096
#define EPSF   1.1920928955078125e-07f

// ---------------------------------------------------------------------------
// WMMA fragment loaders (CDNA5 16x16x32 f16 layouts, CDNA5 ISA)
// A (16xK, row per lane): lane holds M = lane%16; half h -> K = kbase + h +
//   (h>=8 ? 8 : 0) + (lane>=16 ? 8 : 0)  == two contiguous 8-half chunks.
// B (Kx16, col per lane): lane holds N = lane%16; same K order as A:
//   two contiguous 8-half chunks at kbase + (lane>=16 ? 8 : 0) and +16.
// ---------------------------------------------------------------------------
__device__ __forceinline__ v16h load_a16(const _Float16* base, int stride,
                                         int mbase, int kbase, int lane) {
  int m    = mbase + (lane & 15);
  int koff = (lane >> 4) << 3;                   // 0 or 8
  const _Float16* p = base + m * stride + kbase + koff;
  union { v16h v; v8h h[2]; } u;
  u.h[0] = *(const v8h*)p;
  u.h[1] = *(const v8h*)(p + 16);
  return u.v;
}

__device__ __forceinline__ v16h load_b16(const _Float16* base, int stride,
                                         int nbase, int kbase, int lane) {
  int n   = nbase + (lane & 15);
  int klo = (lane >> 4) << 3;                    // 0 or 8
  const _Float16* p = base + n * stride + kbase + klo;
  union { v16h v; v8h h[2]; } u;
  u.h[0] = *(const v8h*)p;
  u.h[1] = *(const v8h*)(p + 16);
  return u.v;
}

// ---------------------------------------------------------------------------
// Prep: convert qkv_w [768x256] and proj_w [256x256] f32 -> f16 (L2 resident)
// ---------------------------------------------------------------------------
__global__ void prep_kernel(const float* __restrict__ qkvw,
                            const float* __restrict__ projw,
                            _Float16* __restrict__ qkvw16,
                            _Float16* __restrict__ projw16) {
  int i = blockIdx.x * blockDim.x + threadIdx.x;
  if (i < 768 * 256) qkvw16[i] = (_Float16)qkvw[i];
  if (i < 256 * 256) projw16[i] = (_Float16)projw[i];
}

// ---------------------------------------------------------------------------
// CPB: bias[h][n][m] = 16*sigmoid( MLP(coords)[REL_IDX[n][m]][h] ), W = 8
// ---------------------------------------------------------------------------
__global__ __launch_bounds__(256)
void cpb_kernel(const float* __restrict__ w1, const float* __restrict__ b1,
                const float* __restrict__ w2, float* __restrict__ biasg) {
  __shared__ float table[225 * 8];
  int tid = threadIdx.x;
  if (tid < 225) {
    int rr = tid / 15, cc = tid % 15;
    float va = (float)(rr - 7), vb = (float)(cc - 7);
    auto xf = [](float v) {
      float g = v * (8.0f / 7.0f);
      float s = (g > 0.0f) ? 1.0f : ((g < 0.0f) ? -1.0f : 0.0f);
      return s * log2f(fabsf(g) + 1.0f) * (1.0f / 3.0f);   // /log2(8)
    };
    float c0 = xf(va), c1 = xf(vb);                         // 3rd channel = 0
    float acc[8] = {0.f,0.f,0.f,0.f,0.f,0.f,0.f,0.f};
    for (int j = 0; j < 512; ++j) {
      float hdn = fmaxf(0.0f, c0 * w1[j * 3] + c1 * w1[j * 3 + 1] + b1[j]);
      #pragma unroll
      for (int hh = 0; hh < 8; ++hh) acc[hh] += w2[hh * 512 + j] * hdn;
    }
    #pragma unroll
    for (int hh = 0; hh < 8; ++hh) table[tid * 8 + hh] = acc[hh];
  }
  __syncthreads();
  for (int idx = tid; idx < 8 * 64 * 64; idx += 256) {
    int hh  = idx >> 12;
    int rem = idx & 4095;
    int n = rem >> 6, m = rem & 63;
    int ridx = ((n >> 3) - (m >> 3) + 7) * 15 + ((n & 7) - (m & 7) + 7);
    float tv = table[ridx * 8 + hh];
    biasg[idx] = 16.0f / (1.0f + __expf(-tv));
  }
}

// ---------------------------------------------------------------------------
// Fused window attention: 1 workgroup = 1 window, 8 waves = 8 heads.
// ---------------------------------------------------------------------------
__global__ __launch_bounds__(256)
void wattn_kernel(const float* __restrict__ x,
                  const float* __restrict__ qkv_b,
                  const float* __restrict__ nqw,
                  const float* __restrict__ nkw,
                  const float* __restrict__ proj_b,
                  const _Float16* __restrict__ qkvw16,
                  const _Float16* __restrict__ projw16,
                  const float* __restrict__ biasg,
                  float* __restrict__ out) {
  __shared__ __align__(16) _Float16 xs[NTOK * DIMC];            // 32KB, x then y
  __shared__ __align__(16) _Float16 qn_s[HEADS][NTOK * HDIM];   // 32KB
  __shared__ __align__(16) _Float16 kn_s[HEADS][NTOK * HDIM];   // 32KB
  __shared__ __align__(16) _Float16 vT_s[HEADS][HDIM * NTOK];   // 32KB
  __shared__ __align__(16) _Float16 p_s [HEADS][NTOK * NTOK];   // 64KB

  const int b    = blockIdx.x;
  const int tid  = threadIdx.x;
  const int h    = tid >> 5;            // wave id == head id
  const int lane = tid & 31;
  const int ln16 = lane & 15;
  const int hi8  = (lane >> 4) << 3;    // D-frag row offset for upper half-wave

  // ---- Stage 0: cooperative load of x tile, f32 -> f16 -------------------
  {
    const float4* xg4 = (const float4*)(x + (size_t)b * NTOK * DIMC);
    #pragma unroll
    for (int it = 0; it < 16; ++it) {
      int j = it * 256 + tid;
      float4 f = xg4[j];
      v4h hv;
      hv[0] = (_Float16)f.x; hv[1] = (_Float16)f.y;
      hv[2] = (_Float16)f.z; hv[3] = (_Float16)f.w;
      *(v4h*)&xs[j * 4] = hv;
    }
  }
  __syncthreads();

  // ---- Stage 1: per-head QKV GEMM (64x256 @ 256x96) + RMSNorm(q,k) -------
  #pragma unroll
  for (int part = 0; part < 3; ++part) {            // 0=q 1=k 2=v
    const int colg = part * DIMC + h * HDIM;        // wave's 32 output cols
    v8f acc[4][2] = {};
    #pragma unroll
    for (int nt = 0; nt < 2; ++nt) {
      for (int ks = 0; ks < 8; ++ks) {
        v16h bfr = load_b16(qkvw16, DIMC, colg + nt * 16, ks * 32, lane);
        if (ks < 7) {   // pull next weight line toward the WGP (global_prefetch_b8)
          __builtin_prefetch(qkvw16 + (colg + nt * 16 + ln16) * DIMC + (ks + 1) * 32, 0, 3);
        }
        #pragma unroll
        for (int mt = 0; mt < 4; ++mt) {
          v16h afr = load_a16(xs, DIMC, mt * 16, ks * 32, lane);
          acc[mt][nt] = __builtin_amdgcn_wmma_f32_16x16x32_f16(
              false, afr, false, bfr, (short)0, acc[mt][nt], false, false);
        }
      }
    }
    float bb0 = qkv_b[colg + ln16];
    float bb1 = qkv_b[colg + 16 + ln16];
    #pragma unroll
    for (int mt = 0; mt < 4; ++mt)
      #pragma unroll
      for (int r = 0; r < 8; ++r) { acc[mt][0][r] += bb0; acc[mt][1][r] += bb1; }

    if (part < 2) {
      const float* nw = (part == 0) ? nqw : nkw;
      _Float16* dst  = (part == 0) ? qn_s[h] : kn_s[h];
      float w0 = nw[ln16], w1 = nw[16 + ln16];
      #pragma unroll
      for (int mt = 0; mt < 4; ++mt) {
        #pragma unroll
        for (int r = 0; r < 8; ++r) {
          float a0 = acc[mt][0][r], a1 = acc[mt][1][r];
          float ss = a0 * a0 + a1 * a1;              // row M lives in a 16-lane half
          ss += __shfl_xor(ss, 1, 32);
          ss += __shfl_xor(ss, 2, 32);
          ss += __shfl_xor(ss, 4, 32);
          ss += __shfl_xor(ss, 8, 32);
          float sc = rsqrtf(ss * (1.0f / HDIM) + EPSF);
          int M = mt * 16 + r + hi8;
          dst[M * HDIM + ln16]      = (_Float16)(a0 * sc * w0);
          dst[M * HDIM + 16 + ln16] = (_Float16)(a1 * sc * w1);
        }
      }
    } else {                                         // v: store transposed
      #pragma unroll
      for (int mt = 0; mt < 4; ++mt)
        #pragma unroll
        for (int nt = 0; nt < 2; ++nt)
          #pragma unroll
          for (int r = 0; r < 8; ++r) {
            int M = mt * 16 + r + hi8;
            vT_s[h][(nt * 16 + ln16) * NTOK + M] = (_Float16)acc[mt][nt][r];
          }
    }
  }
  __syncthreads();   // all x reads done -> xs reusable as y below

  // ---- Stage 2: scores = qn @ kn^T + bias, softmax (wave-private) --------
  const float* bh = biasg + h * NTOK * NTOK;
  #pragma unroll
  for (int mt = 0; mt < 4; ++mt) {
    v16h aq = load_a16(qn_s[h], HDIM, mt * 16, 0, lane);
    v8f sc[4];
    #pragma unroll
    for (int nt = 0; nt < 4; ++nt) {
      v16h bk = load_b16(kn_s[h], HDIM, nt * 16, 0, lane);
      v8f z = {};
      sc[nt] = __builtin_amdgcn_wmma_f32_16x16x32_f16(
          false, aq, false, bk, (short)0, z, false, false);
    }
    #pragma unroll
    for (int r = 0; r < 8; ++r) {
      int M = mt * 16 + r + hi8;
      float s0 = sc[0][r] + bh[M * NTOK + ln16];
      float s1 = sc[1][r] + bh[M * NTOK + 16 + ln16];
      float s2 = sc[2][r] + bh[M * NTOK + 32 + ln16];
      float s3 = sc[3][r] + bh[M * NTOK + 48 + ln16];
      float mx = fmaxf(fmaxf(s0, s1), fmaxf(s2, s3));
      mx = fmaxf(mx, __shfl_xor(mx, 1, 32));
      mx = fmaxf(mx, __shfl_xor(mx, 2, 32));
      mx = fmaxf(mx, __shfl_xor(mx, 4, 32));
      mx = fmaxf(mx, __shfl_xor(mx, 8, 32));
      float e0 = __expf(s0 - mx), e1 = __expf(s1 - mx);
      float e2 = __expf(s2 - mx), e3 = __expf(s3 - mx);
      float sum = e0 + e1 + e2 + e3;
      sum += __shfl_xor(sum, 1, 32);
      sum += __shfl_xor(sum, 2, 32);
      sum += __shfl_xor(sum, 4, 32);
      sum += __shfl_xor(sum, 8, 32);
      float inv = __frcp_rn(sum);
      p_s[h][M * NTOK + ln16]      = (_Float16)(e0 * inv);
      p_s[h][M * NTOK + 16 + ln16] = (_Float16)(e1 * inv);
      p_s[h][M * NTOK + 32 + ln16] = (_Float16)(e2 * inv);
      p_s[h][M * NTOK + 48 + ln16] = (_Float16)(e3 * inv);
    }
  }

  // ---- Stage 3: y = P @ V  (64x64 @ 64x32), write y into xs --------------
  #pragma unroll
  for (int mt = 0; mt < 4; ++mt) {
    #pragma unroll
    for (int nt = 0; nt < 2; ++nt) {
      v8f acc = {};
      #pragma unroll
      for (int ks = 0; ks < 2; ++ks) {
        v16h ap = load_a16(p_s[h], NTOK, mt * 16, ks * 32, lane);
        v16h bv = load_b16(vT_s[h], NTOK, nt * 16, ks * 32, lane);
        acc = __builtin_amdgcn_wmma_f32_16x16x32_f16(
            false, ap, false, bv, (short)0, acc, false, false);
      }
      #pragma unroll
      for (int r = 0; r < 8; ++r) {
        int M = mt * 16 + r + hi8;
        xs[M * DIMC + h * HDIM + nt * 16 + ln16] = (_Float16)acc[r];
      }
    }
  }
  __syncthreads();   // full y tile visible to all waves

  // ---- Stage 4: out = y @ proj_w^T + proj_b  (wave covers 32 cols) -------
  // B fragment hoisted out of the mt loop: each proj weight fragment is now
  // fetched exactly once per wave (was 4x).
  #pragma unroll
  for (int nt = 0; nt < 2; ++nt) {
    const int colb = h * 32 + nt * 16;
    float pb = proj_b[colb + ln16];
    v8f acc[4] = {};
    for (int ks = 0; ks < 8; ++ks) {
      v16h bw = load_b16(projw16, DIMC, colb, ks * 32, lane);
      if (ks < 7) {
        __builtin_prefetch(projw16 + (colb + ln16) * DIMC + (ks + 1) * 32, 0, 3);
      }
      #pragma unroll
      for (int mt = 0; mt < 4; ++mt) {
        v16h ay = load_a16(xs, DIMC, mt * 16, ks * 32, lane);
        acc[mt] = __builtin_amdgcn_wmma_f32_16x16x32_f16(
            false, ay, false, bw, (short)0, acc[mt], false, false);
      }
    }
    float* og = out + (size_t)b * NTOK * DIMC;
    #pragma unroll
    for (int mt = 0; mt < 4; ++mt)
      #pragma unroll
      for (int r = 0; r < 8; ++r) {
        int M = mt * 16 + r + hi8;
        og[M * DIMC + colb + ln16] = acc[mt][r] + pb;
      }
  }
}

// ---------------------------------------------------------------------------
extern "C" void kernel_launch(void* const* d_in, const int* in_sizes, int n_in,
                              void* d_out, int out_size, void* d_ws, size_t ws_size,
                              hipStream_t stream) {
  const float* x      = (const float*)d_in[0];
  const float* qkv_w  = (const float*)d_in[1];
  const float* qkv_b  = (const float*)d_in[2];
  const float* cpb_w1 = (const float*)d_in[3];
  const float* cpb_b1 = (const float*)d_in[4];
  const float* cpb_w2 = (const float*)d_in[5];
  const float* nqw    = (const float*)d_in[6];
  const float* nkw    = (const float*)d_in[7];
  const float* projw  = (const float*)d_in[8];
  const float* projb  = (const float*)d_in[9];
  // d_in[10] = num_dim (always 2)

  char* ws = (char*)d_ws;
  float*    biasg   = (float*)ws;                           // 8*64*64*4  = 128KB
  _Float16* qkvw16  = (_Float16*)(ws + 131072);             // 768*256*2  = 384KB
  _Float16* projw16 = (_Float16*)(ws + 131072 + 393216);    // 256*256*2  = 128KB

  prep_kernel<<<768, 256, 0, stream>>>(qkv_w, projw, qkvw16, projw16);
  cpb_kernel<<<1, 256, 0, stream>>>(cpb_w1, cpb_b1, cpb_w2, biasg);
  wattn_kernel<<<NWIN, 256, 0, stream>>>(x, qkv_b, nqw, nkw, projb,
                                         qkvw16, projw16, biasg, (float*)d_out);
}
